// MultiHeadAttention_69655779607087
// MI455X (gfx1250) — matrix operand order repaired and checked by reading, not compiled, not run on hardware
//
#include <hip/hip_runtime.h>
#include <hip/hip_bf16.h>

// ---------------------------------------------------------------------------
// MultiHeadAttention forward for MI455X (gfx1250, wave32, WMMA bf16 + TDM)
//   H=16, D_MODEL=1024, D_QK=D_V=64, B=4, L=2048
// Pipeline: [1] QKV proj (f32 x, f32 W -> bf16 q/k/v in ws)
//           [2] flash attention per (b,h,qtile16), K/V tiles streamed into
//               LDS by the Tensor Data Mover (double buffered, TENSORcnt)
//           [3] out projection (A tile TDM-staged into LDS, bf16 x Wout -> f32)
// All matmuls: v_wmma_f32_16x16x32_bf16, f32 accumulation, f32 softmax.
// ---------------------------------------------------------------------------

typedef __bf16 bf16_t;
typedef bf16_t v16bf __attribute__((ext_vector_type(16)));
typedef float  v8f   __attribute__((ext_vector_type(8)));
typedef unsigned int u32x4 __attribute__((ext_vector_type(4)));
typedef int          i32x8 __attribute__((ext_vector_type(8)));
typedef int          i32x4 __attribute__((ext_vector_type(4)));

#define NHEAD   16
#define DMODEL  1024
#define DHEAD   64
#define NBATCH  4
#define SEQLEN  2048
#define BL      (NBATCH * SEQLEN)   // 8192 rows

static __device__ __forceinline__ unsigned short f2bf_bits(float f) {
    unsigned int u = __builtin_bit_cast(unsigned int, f);
    u += 0x7FFFu + ((u >> 16) & 1u);          // round-to-nearest-even
    return (unsigned short)(u >> 16);
}
static __device__ __forceinline__ bf16_t bits2bf(unsigned short h) {
    return __builtin_bit_cast(bf16_t, h);
}
static __device__ __forceinline__ bf16_t f2bf(float f) {
    return bits2bf(f2bf_bits(f));
}
static __device__ __forceinline__ v8f zero8() {
    v8f z;
#pragma unroll
    for (int i = 0; i < 8; ++i) z[i] = 0.0f;
    return z;
}
// A-fragment K index for 16-bit 16x32 A matrix (ISA 7.12.2):
// vgpr v<4: K = half*8 + 2v (+j) ; v>=4: K = 16 + half*8 + 2(v-4) (+j)
static __device__ __forceinline__ int a_kidx(int v, int half) {
    return (v < 4) ? (half * 8 + 2 * v) : (16 + half * 8 + 2 * (v - 4));
}
// Flat shared-memory pointer -> LDS byte address (ISA 10.2: LDS aperture
// addresses are addr[31:0]).
static __device__ __forceinline__ unsigned lds_addr_of(const void* p) {
    return (unsigned)(unsigned long long)p;
}

// ---------------------------------------------------------------------------
// Tensor Data Mover: 2D tile load Global -> LDS (ISA ch.8, D# groups 0/1).
// 16-bit elements. tile_d0 = contiguous elems per row, tile_d1 = rows,
// stride0 = row stride in elements. Tracked with TENSORcnt.
// ---------------------------------------------------------------------------
static __device__ __forceinline__ void tdm_load_2d(
    const void* gptr, unsigned lds_byte,
    unsigned tensor_d0, unsigned tensor_d1, unsigned long long stride0,
    unsigned tile_d0, unsigned tile_d1)
{
    unsigned long long ga = (unsigned long long)gptr;
    u32x4 g0;
    g0.x = 1u;                                           // count=1, user D#
    g0.y = lds_byte;                                     // lds_addr [63:32]
    g0.z = (unsigned)(ga & 0xffffffffu);                 // global_addr lo
    g0.w = (unsigned)((ga >> 32) & 0x1ffffffu)           // global_addr hi
         | (2u << 30);                                   // type = 2 (image)
    i32x8 g1;
    g1[0] = 0x10000;                                     // data_size=1 (2B)
    g1[1] = (int)((tensor_d0 & 0xffffu) << 16);          // tensor_dim0 lo16
    g1[2] = (int)((tensor_d0 >> 16) | ((tensor_d1 & 0xffffu) << 16));
    g1[3] = (int)((tensor_d1 >> 16) | (tile_d0 << 16));  // tile_dim0
    g1[4] = (int)(tile_d1 & 0xffffu);                    // tile_dim1; dim2=0
    g1[5] = (int)(stride0 & 0xffffffffu);                // dim0 stride lo
    g1[6] = (int)((stride0 >> 32) & 0xffffu);            // dim0 stride hi
    g1[7] = 0;
    i32x4 gz; gz[0] = 0; gz[1] = 0; gz[2] = 0; gz[3] = 0;
    asm volatile("tensor_load_to_lds %0, %1, %2, %3"
                 :: "s"(g0), "s"(g1), "s"(gz), "s"(gz) : "memory");
}

// ---------------------------------------------------------------------------
// Kernel 1: QKV projection.  grid(512, 2, 3), block 256 (8 waves).
// wave w of block (x=mt, y=hg, z=mat) computes 16x64 tile of head hg*8+w.
// (x is f32 and needs conversion, so this stages via VALU, not TDM.)
// ---------------------------------------------------------------------------
__global__ __launch_bounds__(256) void qkv_proj_kernel(
    const float* __restrict__ xq, const float* __restrict__ xk,
    const float* __restrict__ xv,
    const float* __restrict__ Wq, const float* __restrict__ Wk,
    const float* __restrict__ Wv,
    unsigned short* __restrict__ qo, unsigned short* __restrict__ ko,
    unsigned short* __restrict__ vo)
{
    const float* x; const float* W; unsigned short* out;
    if (blockIdx.z == 0)      { x = xq; W = Wq; out = qo; }
    else if (blockIdx.z == 1) { x = xk; W = Wk; out = ko; }
    else                      { x = xv; W = Wv; out = vo; }

    const int mt   = blockIdx.x;                    // 16-row tile of BL
    const int head = blockIdx.y * 8 + (threadIdx.x >> 5);
    const int lane = threadIdx.x & 31;
    const int half = lane >> 4;
    const int l16  = lane & 15;
    const int row0 = mt * 16;
    const float* Wb = W + head * DMODEL * DHEAD;    // [1024 x 64] row-major

    __shared__ unsigned short As[16][34];           // bf16 x-tile, padded

    v8f acc[4];
#pragma unroll
    for (int t = 0; t < 4; ++t) acc[t] = zero8();

    for (int kk = 0; kk < DMODEL; kk += 32) {
        // cooperative stage: 16x32 x-tile, f32 -> bf16
#pragma unroll
        for (int rep = 0; rep < 2; ++rep) {
            int i = threadIdx.x + rep * 256;
            int r = i >> 5, c = i & 31;
            As[r][c] = f2bf_bits(x[(row0 + r) * DMODEL + kk + c]);
        }
        __syncthreads();

        v16bf afrag;
#pragma unroll
        for (int v = 0; v < 8; ++v) {
            int k0 = half * 16 + 2 * v;
            afrag[2 * v]     = bits2bf(As[l16][k0]);
            afrag[2 * v + 1] = bits2bf(As[l16][k0 + 1]);
        }
#pragma unroll
        for (int t = 0; t < 4; ++t) {
            v16bf bfrag;
#pragma unroll
            for (int v = 0; v < 8; ++v) {
                int d0 = kk + half * 16 + 2 * v;    // B: lane%16=N, half picks K+16
                bfrag[2 * v]     = f2bf(Wb[d0 * DHEAD + t * 16 + l16]);
                bfrag[2 * v + 1] = f2bf(Wb[(d0 + 1) * DHEAD + t * 16 + l16]);
            }
            acc[t] = __builtin_amdgcn_wmma_f32_16x16x32_bf16(
                false, afrag, false, bfrag, (short)0, acc[t], false, false);
        }
        __syncthreads();
    }

    // store to [B, H, L, 64] bf16
#pragma unroll
    for (int r = 0; r < 8; ++r) {
        int row = row0 + r + 8 * half;              // C/D: M = vgpr + 8*half
        int b = row >> 11, l = row & (SEQLEN - 1);
        int base = ((b * NHEAD + head) * SEQLEN + l) * DHEAD;
#pragma unroll
        for (int t = 0; t < 4; ++t)
            out[base + t * 16 + l16] = f2bf_bits(acc[t][r]);
    }
}

// ---------------------------------------------------------------------------
// Kernel 2: flash attention.  grid(128, 16, 4) = (qtile, head, batch), 1 wave.
// K/V 32x64 bf16 tiles streamed Global->LDS by TDM, double buffered on
// TENSORcnt; fragments gathered from LDS; online softmax; P relayout via LDS.
// ---------------------------------------------------------------------------
__global__ __launch_bounds__(32) void flash_attn_kernel(
    const unsigned short* __restrict__ q, const unsigned short* __restrict__ k,
    const unsigned short* __restrict__ v, unsigned short* __restrict__ heads)
{
    const int qt = blockIdx.x, h = blockIdx.y, b = blockIdx.z;
    const int lane = threadIdx.x & 31;
    const int half = lane >> 4;
    const int l16  = lane & 15;

    const unsigned short* Qb = q + (((b * NHEAD + h) * SEQLEN) + qt * 16) * DHEAD;
    const unsigned short* Kb = k + ((b * NHEAD + h) * SEQLEN) * DHEAD;
    const unsigned short* Vb = v + ((b * NHEAD + h) * SEQLEN) * DHEAD;

    __shared__ unsigned short Ks[2][32][DHEAD];     // TDM double buffers (8 KB)
    __shared__ unsigned short Vs[2][32][DHEAD];     // (8 KB)
    __shared__ unsigned short Ps[16][34];           // P tile, padded

    // Q A-fragments for d-chunks [0,32) and [32,64)
    v16bf qf[2];
#pragma unroll
    for (int c = 0; c < 2; ++c)
#pragma unroll
        for (int vv = 0; vv < 8; ++vv) {
            int k0 = c * 32 + half * 16 + 2 * vv;
            qf[c][2 * vv]     = bits2bf(Qb[l16 * DHEAD + k0]);
            qf[c][2 * vv + 1] = bits2bf(Qb[l16 * DHEAD + k0 + 1]);
        }

    v8f acc[4];
#pragma unroll
    for (int t = 0; t < 4; ++t) acc[t] = zero8();
    float mrow[8], lrow[8];
#pragma unroll
    for (int r = 0; r < 8; ++r) { mrow[r] = -1e30f; lrow[r] = 0.0f; }

    // prologue: TDM kick for tile 0 into buffer 0
    tdm_load_2d(Kb, lds_addr_of(&Ks[0][0][0]), DHEAD, SEQLEN, DHEAD, DHEAD, 32);
    tdm_load_2d(Vb, lds_addr_of(&Vs[0][0][0]), DHEAD, SEQLEN, DHEAD, DHEAD, 32);

    for (int s0 = 0; s0 < SEQLEN; s0 += 32) {
        const int buf = (s0 >> 5) & 1;
        if (s0 + 32 < SEQLEN) {                     // stream next tile
            tdm_load_2d(Kb + (s0 + 32) * DHEAD, lds_addr_of(&Ks[buf ^ 1][0][0]),
                        DHEAD, SEQLEN, DHEAD, DHEAD, 32);
            tdm_load_2d(Vb + (s0 + 32) * DHEAD, lds_addr_of(&Vs[buf ^ 1][0][0]),
                        DHEAD, SEQLEN, DHEAD, DHEAD, 32);
            __builtin_amdgcn_s_wait_tensorcnt(2);   // current tile landed (in order)
        } else {
            __builtin_amdgcn_s_wait_tensorcnt(0);
        }

        // scores: two 16x16 C tiles over kv columns [s0, s0+32)
        v8f sc[2];
#pragma unroll
        for (int nt = 0; nt < 2; ++nt) {
            sc[nt] = zero8();
            int srow = nt * 16 + l16;               // kv row within tile
#pragma unroll
            for (int c = 0; c < 2; ++c) {
                v16bf kf;                           // B = K^T : B[d][s] = K[s][d]
#pragma unroll
                for (int vv = 0; vv < 8; ++vv) {
                    int d0 = c * 32 + half * 16 + 2 * vv;
                    kf[2 * vv]     = bits2bf(Ks[buf][srow][d0]);
                    kf[2 * vv + 1] = bits2bf(Ks[buf][srow][d0 + 1]);
                }
                sc[nt] = __builtin_amdgcn_wmma_f32_16x16x32_bf16(
                    false, qf[c], false, kf, (short)0, sc[nt], false, false);
            }
        }
        // online softmax (rows live across 16-lane halves of the wave)
#pragma unroll
        for (int r = 0; r < 8; ++r) {
            float a0 = sc[0][r] * 0.125f;           // 1/sqrt(64)
            float a1 = sc[1][r] * 0.125f;
            float mx = fmaxf(a0, a1);
#pragma unroll
            for (int off = 1; off < 16; off <<= 1)
                mx = fmaxf(mx, __shfl_xor(mx, off, 16));
            float mnew = fmaxf(mrow[r], mx);
            float p0 = __expf(a0 - mnew);
            float p1 = __expf(a1 - mnew);
            float rs = p0 + p1;
#pragma unroll
            for (int off = 1; off < 16; off <<= 1)
                rs += __shfl_xor(rs, off, 16);
            float alpha = __expf(mrow[r] - mnew);
            lrow[r] = lrow[r] * alpha + rs;
            mrow[r] = mnew;
#pragma unroll
            for (int t = 0; t < 4; ++t) acc[t][r] *= alpha;
            int m = r + 8 * half;                   // C-layout row
            Ps[m][l16]      = f2bf_bits(p0);
            Ps[m][16 + l16] = f2bf_bits(p1);
        }
        __syncthreads();                            // C-layout -> A-layout handoff
        v16bf pf;
#pragma unroll
        for (int vv = 0; vv < 8; ++vv) {
            int k0 = half * 16 + 2 * vv;
            pf[2 * vv]     = bits2bf(Ps[l16][k0]);
            pf[2 * vv + 1] = bits2bf(Ps[l16][k0 + 1]);
        }
        // acc += P (16x32) x V (32x64), V fragments from TDM-staged LDS tile
#pragma unroll
        for (int t = 0; t < 4; ++t) {
            v16bf vf;
#pragma unroll
            for (int vv = 0; vv < 8; ++vv) {
                int kr = half * 16 + 2 * vv;
                vf[2 * vv]     = bits2bf(Vs[buf][kr][t * 16 + l16]);
                vf[2 * vv + 1] = bits2bf(Vs[buf][kr + 1][t * 16 + l16]);
            }
            acc[t] = __builtin_amdgcn_wmma_f32_16x16x32_bf16(
                false, pf, false, vf, (short)0, acc[t], false, false);
        }
        __syncthreads();
    }

    // normalize and store concat-heads bf16: [B, L, H*64]
#pragma unroll
    for (int r = 0; r < 8; ++r) {
        float invl = 1.0f / lrow[r];
        int row = qt * 16 + r + 8 * half;
        int base = (b * SEQLEN + row) * (NHEAD * DHEAD) + h * DHEAD;
#pragma unroll
        for (int t = 0; t < 4; ++t)
            heads[base + t * 16 + l16] = f2bf_bits(acc[t][r] * invl);
    }
}

// ---------------------------------------------------------------------------
// Kernel 3: output projection.  heads[8192,1024](bf16) x Wout[1024,1024](f32)
// grid(512, 8), block 256 (8 waves); wave w -> N tile blockIdx.y*8+w.
// A tile (16x32 bf16) is TDM-staged into LDS by wave 0, published by barrier.
// ---------------------------------------------------------------------------
__global__ __launch_bounds__(256) void out_proj_kernel(
    const unsigned short* __restrict__ heads, const float* __restrict__ Wout,
    float* __restrict__ out)
{
    const int mt    = blockIdx.x;
    const int ntile = blockIdx.y * 8 + (threadIdx.x >> 5);
    const int lane  = threadIdx.x & 31;
    const int half  = lane >> 4;
    const int l16   = lane & 15;
    const int row0  = mt * 16;

    __shared__ unsigned short As[16][32];           // contiguous for TDM

    v8f acc = zero8();
    for (int kk = 0; kk < NHEAD * DHEAD; kk += 32) {
        if ((threadIdx.x >> 5) == 0) {              // wave 0 drives the TDM
            tdm_load_2d(heads + row0 * DMODEL + kk, lds_addr_of(&As[0][0]),
                        DMODEL, BL, DMODEL, 32, 16);
            __builtin_amdgcn_s_wait_tensorcnt(0);
        }
        __syncthreads();

        v16bf afrag;
#pragma unroll
        for (int v = 0; v < 8; ++v) {
            int k0 = half * 16 + 2 * v;
            afrag[2 * v]     = bits2bf(As[l16][k0]);
            afrag[2 * v + 1] = bits2bf(As[l16][k0 + 1]);
        }
        v16bf bfrag;
#pragma unroll
        for (int v = 0; v < 8; ++v) {
            int k0 = kk + half * 16 + 2 * v;
            bfrag[2 * v]     = f2bf(Wout[k0 * DMODEL + ntile * 16 + l16]);
            bfrag[2 * v + 1] = f2bf(Wout[(k0 + 1) * DMODEL + ntile * 16 + l16]);
        }
        acc = __builtin_amdgcn_wmma_f32_16x16x32_bf16(
            false, afrag, false, bfrag, (short)0, acc, false, false);
        __syncthreads();
    }
#pragma unroll
    for (int r = 0; r < 8; ++r)
        out[(row0 + r + 8 * half) * DMODEL + ntile * 16 + l16] = acc[r];
}

// ---------------------------------------------------------------------------
extern "C" void kernel_launch(void* const* d_in, const int* in_sizes, int n_in,
                              void* d_out, int out_size, void* d_ws, size_t ws_size,
                              hipStream_t stream) {
    (void)in_sizes; (void)n_in; (void)out_size; (void)ws_size;
    const float* xq   = (const float*)d_in[0];
    const float* xk   = (const float*)d_in[1];
    const float* xv   = (const float*)d_in[2];
    const float* Wq   = (const float*)d_in[3];
    const float* Wk   = (const float*)d_in[4];
    const float* Wv   = (const float*)d_in[5];
    const float* Wout = (const float*)d_in[6];
    float* out        = (float*)d_out;

    // workspace layout (bf16 stored as u16): q,k,v [B,H,L,64]; heads [B,L,1024]
    const size_t QKV_ELEMS = (size_t)NBATCH * NHEAD * SEQLEN * DHEAD;  // 8.39M
    unsigned short* q_ws = (unsigned short*)d_ws;
    unsigned short* k_ws = q_ws + QKV_ELEMS;
    unsigned short* v_ws = k_ws + QKV_ELEMS;
    unsigned short* h_ws = v_ws + QKV_ELEMS;                           // 16 MB
    // total ws use: 64 MB

    qkv_proj_kernel<<<dim3(BL / 16, 2, 3), 256, 0, stream>>>(
        xq, xk, xv, Wq, Wk, Wv, q_ws, k_ws, v_ws);
    flash_attn_kernel<<<dim3(SEQLEN / 16, NHEAD, NBATCH), 32, 0, stream>>>(
        q_ws, k_ws, v_ws, h_ws);
    out_proj_kernel<<<dim3(BL / 16, 8), 256, 0, stream>>>(h_ws, Wout, out);
}
